// MemorizingLlamaDecoderLayer_61314953118497
// MI455X (gfx1250) — compile-verified, matrix operand order checked by reading, not run on hardware
//
#include <hip/hip_runtime.h>
#include <hip/hip_bf16.h>

// ---------------- problem dims (match reference) ----------------
constexpr int Bc = 2, Sc = 1024, Dc = 2048, Hc = 16, Fc = 8192, Mc = 8192;
constexpr int HDc = Dc / Hc;           // 128
constexpr long BSc = (long)Bc * Sc;    // 2048 rows
constexpr float RMS_EPSc  = 1e-6f;
constexpr float NORM_EPSc = 1e-4f;
constexpr float ROPE_THETAc = 10000.0f;

typedef __attribute__((ext_vector_type(16))) __bf16 v16bf;
typedef __attribute__((ext_vector_type(8)))  float  v8f;

// ---------------- TDM availability ----------------
#if defined(__has_builtin)
#if __has_builtin(__builtin_amdgcn_tensor_load_to_lds) && \
    __has_builtin(__builtin_amdgcn_s_wait_tensorcnt)
#define USE_TDM 1
#endif
#endif
#ifndef USE_TDM
#define USE_TDM 0
#endif

typedef __attribute__((ext_vector_type(4))) unsigned int tdm_v4u;
typedef __attribute__((ext_vector_type(8))) int tdm_v8i;
typedef __attribute__((ext_vector_type(4))) int tdm_v4i;

// ---------------- bf16 helpers (bit-exact RNE) ----------------
__device__ __forceinline__ unsigned short f2bf(float f) {
  unsigned int u = __builtin_bit_cast(unsigned int, f);
  unsigned int r = u + 0x7FFFu + ((u >> 16) & 1u);
  return (unsigned short)(r >> 16);
}
__device__ __forceinline__ __bf16 bfbits(unsigned short u) {
  return __builtin_bit_cast(__bf16, u);
}

// ---------------- WMMA fragment loaders --------------------------------
// ISA 7.12.2: 16-bit A matrix 16x32 (MxK). Lane L(0..15): row M=L, holds
// K in {0..7,16..23}; lane L+16: row M=L, K in {8..15,24..31}.
// Element h of v16bf maps to k = h + (h&8) + ((lane&16)>>1).
__device__ __forceinline__ v16bf load_a_frag(const unsigned short* A, int ld) {
  int lane = threadIdx.x & 31;
  int kh = (lane & 16) >> 1;
  const unsigned short* p = A + (long)(lane & 15) * ld;
  v16bf a;
#pragma unroll
  for (int h = 0; h < 16; ++h) a[h] = bfbits(p[h + (h & 8) + kh]);
  return a;
}
// B matrix 32x16 (KxN) from a row-major [K x N] source: element (k,n).
__device__ __forceinline__ v16bf load_b_frag(const unsigned short* Bw, int ld) {
  int lane = threadIdx.x & 31;
  int kh = (lane & 16) >> 1;
  int n = lane & 15;
  v16bf b;
#pragma unroll
  for (int h = 0; h < 16; ++h) {
    int k = h + (h & 8) + kh;
    b[h] = bfbits(Bw[(long)k * ld + n]);
  }
  return b;
}
// B matrix 32x16 where the source is row-major [N x K]: element (k,n)=src[n*ld+k].
__device__ __forceinline__ v16bf load_bt_frag(const unsigned short* Bt, int ld) {
  int lane = threadIdx.x & 31;
  int kh = (lane & 16) >> 1;
  const unsigned short* p = Bt + (long)(lane & 15) * ld;
  v16bf b;
#pragma unroll
  for (int h = 0; h < 16; ++h) b[h] = bfbits(p[h + (h & 8) + kh]);
  return b;
}
__device__ __forceinline__ v8f v8f_zero() {
  v8f z = {0.f, 0.f, 0.f, 0.f, 0.f, 0.f, 0.f, 0.f};
  return z;
}
#define WMMA_BF16(a, b, c) \
  __builtin_amdgcn_wmma_f32_16x16x32_bf16(false, (a), false, (b), (short)0, (c), false, false)

// ---------------- TDM 2D tile load: global (bf16, row-major) -> LDS -------
// D# per ISA 8.3/8.4: group0 {count=1 | lds_addr | global_addr | type=2},
// group1 {data_size=1(2B) | tensor_dim0/1 | tile_dim0/1 | dim0_stride}.
__device__ __forceinline__ void tdm_load_2d(const void* gptr, unsigned lds_byte,
                                            unsigned tile_x, unsigned tile_y,
                                            unsigned long long stride0,
                                            unsigned long long tensor_x,
                                            unsigned long long tensor_y) {
#if USE_TDM
  unsigned long long ga = (unsigned long long)gptr;
  tdm_v4u g0;
  g0[0] = 1u;                                      // count=1 (user descriptor)
  g0[1] = lds_byte;                                // lds_addr
  g0[2] = (unsigned)(ga & 0xFFFFFFFFu);            // global_addr[31:0]
  g0[3] = (unsigned)((ga >> 32) & 0x1FFFFFFu) | (2u << 30);  // [56:32] | type=2
  tdm_v8i g1;
  g1[0] = (int)(1u << 16);                         // data_size = 1 -> 2 bytes
  g1[1] = (int)((tensor_x & 0xFFFFull) << 16);     // tensor_dim0[15:0] @ 63:48
  g1[2] = (int)(((tensor_x >> 16) & 0xFFFFull) |   // tensor_dim0[31:16]
                ((tensor_y & 0xFFFFull) << 16));   // tensor_dim1[15:0]
  g1[3] = (int)(((tensor_y >> 16) & 0xFFFFull) |   // tensor_dim1[31:16]
                ((unsigned long long)(tile_x & 0xFFFFu) << 16));  // tile_dim0
  g1[4] = (int)(tile_y & 0xFFFFu);                 // tile_dim1 (tile_dim2=0)
  g1[5] = (int)(stride0 & 0xFFFFFFFFull);          // dim0_stride[31:0]
  g1[6] = (int)((stride0 >> 32) & 0xFFFFull);      // dim0_stride[47:32]
  g1[7] = 0;
  tdm_v4i gz = {0, 0, 0, 0};
#if __clang_major__ >= 23
  tdm_v8i gz8 = {0, 0, 0, 0, 0, 0, 0, 0};
  __builtin_amdgcn_tensor_load_to_lds(g0, g1, gz, gz, gz8, 0);
#else
  __builtin_amdgcn_tensor_load_to_lds(g0, g1, gz, gz, 0);
#endif
#else
  (void)gptr; (void)lds_byte; (void)tile_x; (void)tile_y;
  (void)stride0; (void)tensor_x; (void)tensor_y;
#endif
}
__device__ __forceinline__ unsigned lds_addr_of(const void* p) {
  // flat->LDS aperture: low 32 bits of generic address are the LDS byte addr
  return (unsigned)(unsigned long long)p;
}

// ---------------- fp32 -> bf16 conversion ----------------
__global__ void cvt_bf16_kernel(const float* __restrict__ src,
                                unsigned short* __restrict__ dst, long n) {
  long i = (long)blockIdx.x * blockDim.x + threadIdx.x;
  long stride = (long)gridDim.x * blockDim.x;
  for (; i < n; i += stride) dst[i] = f2bf(src[i]);
}

// ---------------- kNN retrieval: scores = hs @ mem^T, per-row argmax ----
__global__ __launch_bounds__(256) void knn_argmax_kernel(
    const unsigned short* __restrict__ hs_bf,
    const unsigned short* __restrict__ mem_bf, int* __restrict__ nn_idx) {
  __shared__ float red_v[8][16];
  __shared__ int   red_i[8][16];
  int wave = threadIdx.x >> 5, lane = threadIdx.x & 31;
  int n = lane & 15, rb = (lane >> 4) << 3;
  const unsigned short* Abase = hs_bf + (long)blockIdx.x * 16 * Dc;

  float best_v[8];
  int   best_i[8];
#pragma unroll
  for (int r = 0; r < 8; ++r) { best_v[r] = -3e38f; best_i[r] = 0; }

  for (int mt = wave; mt < Mc / 16; mt += 8) {
    v8f acc = v8f_zero();
    const unsigned short* Bbase = mem_bf + (long)mt * 16 * Dc;
    for (int kk = 0; kk < Dc; kk += 32) {
      v16bf a = load_a_frag(Abase + kk, Dc);
      v16bf b = load_bt_frag(Bbase + kk, Dc);  // scores need mem^T
      acc = WMMA_BF16(a, b, acc);
    }
    int ci = mt * 16 + n;
#pragma unroll
    for (int r = 0; r < 8; ++r) {
      float v = acc[r];
      if (v > best_v[r]) { best_v[r] = v; best_i[r] = ci; }
    }
  }
#pragma unroll
  for (int r = 0; r < 8; ++r) {
    float v = best_v[r]; int i = best_i[r];
#pragma unroll
    for (int mask = 1; mask <= 8; mask <<= 1) {
      float ov = __shfl_xor(v, mask, 32);
      int   oi = __shfl_xor(i, mask, 32);
      if (ov > v || (ov == v && oi < i)) { v = ov; i = oi; }
    }
    best_v[r] = v; best_i[r] = i;
  }
  if ((lane & 15) == 0) {
#pragma unroll
    for (int r = 0; r < 8; ++r) { red_v[wave][rb + r] = best_v[r]; red_i[wave][rb + r] = best_i[r]; }
  }
  __syncthreads();
  if (threadIdx.x < 16) {
    float v = red_v[0][threadIdx.x]; int i = red_i[0][threadIdx.x];
#pragma unroll
    for (int w = 1; w < 8; ++w) {
      float ov = red_v[w][threadIdx.x]; int oi = red_i[w][threadIdx.x];
      if (ov > v || (ov == v && oi < i)) { v = ov; i = oi; }
    }
    nn_idx[blockIdx.x * 16 + threadIdx.x] = i;
  }
}

// ---------------- block reduction helper ----------------
__device__ __forceinline__ float block_reduce_sum(float v, float* red) {
  int tid = threadIdx.x;
  red[tid] = v;
  __syncthreads();
  for (int s = 128; s > 0; s >>= 1) {
    if (tid < s) red[tid] += red[tid + s];
    __syncthreads();
  }
  float r = red[0];
  __syncthreads();
  return r;
}

// ---------------- normalize + gated mix + RMSNorm-1 ----------------
__global__ __launch_bounds__(256) void mix_rms_kernel(
    const float* __restrict__ hs, const float* __restrict__ memb,
    const int* __restrict__ nn_idx, const float* __restrict__ gate_logit,
    const float* __restrict__ rms1_w, float* __restrict__ merged,
    unsigned short* __restrict__ x_bf) {
  __shared__ float red[256];
  int row = blockIdx.x, tid = threadIdx.x;
  const float* hrow = hs + (long)row * Dc;
  const float* mrow = memb + (long)nn_idx[row] * Dc;
  float hv[8], mv[8], mg[8];
  float sh = 0.f, sm = 0.f;
#pragma unroll
  for (int j = 0; j < 8; ++j) {
    int d = tid + j * 256;
    hv[j] = hrow[d]; mv[j] = mrow[d];
    sh += hv[j] * hv[j]; sm += mv[j] * mv[j];
  }
  float norm_h = sqrtf(block_reduce_sum(sh, red)) + NORM_EPSc;
  float norm_m = sqrtf(block_reduce_sum(sm, red)) + NORM_EPSc;
  float s2 = 0.f;
#pragma unroll
  for (int j = 0; j < 8; ++j) {
    int d = tid + j * 256;
    float g = 1.f / (1.f + expf(-gate_logit[d]));
    mg[j] = (g * hv[j] / norm_h + (1.f - g) * mv[j] / norm_m) * norm_h;
    merged[(long)row * Dc + d] = mg[j];
    s2 += mg[j] * mg[j];
  }
  float var = block_reduce_sum(s2, red) / (float)Dc;
  float rinv = rsqrtf(var + RMS_EPSc);
#pragma unroll
  for (int j = 0; j < 8; ++j) {
    int d = tid + j * 256;
    x_bf[(long)row * Dc + d] = f2bf(mg[j] * rinv * rms1_w[d]);
  }
}

// ---------------- RMSNorm-2 ----------------
__global__ __launch_bounds__(256) void rms2_kernel(const float* __restrict__ x,
                                                   const float* __restrict__ w,
                                                   unsigned short* __restrict__ y_bf) {
  __shared__ float red[256];
  int row = blockIdx.x, tid = threadIdx.x;
  const float* xr = x + (long)row * Dc;
  float xv[8]; float s = 0.f;
#pragma unroll
  for (int j = 0; j < 8; ++j) {
    int d = tid + j * 256;
    xv[j] = xr[d]; s += xv[j] * xv[j];
  }
  float rinv = rsqrtf(block_reduce_sum(s, red) / (float)Dc + RMS_EPSc);
#pragma unroll
  for (int j = 0; j < 8; ++j) {
    int d = tid + j * 256;
    y_bf[(long)row * Dc + d] = f2bf(xv[j] * rinv * w[d]);
  }
}

// ---------------- LDS-staged, double-buffered bf16 WMMA GEMM --------------
// Block tile 128(M) x 64(N); 8 waves as 4x2, each wave computes 32x32
// (2x2 WMMA tiles, 4 wmma per 32-wide K step). K-slices of A and B are
// streamed into ping-pong LDS buffers by the Tensor Data Mover: issue tiles
// for step k+1, then s_wait_tensorcnt(2) so only the in-flight pair remains
// outstanding (TDM ops of one wave complete in order), compute step k.
// epi: 0 = C=acc ; 1 = C=acc+extra ; 2 = out_bf = silu(extra)*acc (bf16)
__global__ __launch_bounds__(256) void gemm_bf16_wmma_kernel(
    const unsigned short* __restrict__ A, const unsigned short* __restrict__ Bw,
    float* __restrict__ C, const float* __restrict__ extra,
    unsigned short* __restrict__ out_bf, int Mr, int K, int N, int epi) {
  __shared__ alignas(16) unsigned short a_sh[2][128 * 32];  // [m][k] 8 KB x2
  __shared__ alignas(16) unsigned short b_sh[2][32 * 64];   // [k][n] 4 KB x2
  int wave = threadIdx.x >> 5, lane = threadIdx.x & 31;
  int mw = wave >> 1, nw = wave & 1;
  int nblk = N >> 6;
  int bm = blockIdx.x / nblk, bn = blockIdx.x % nblk;
  int nk = K >> 5;

  auto issue_tiles = [&](int k, int buf) {
    int kk = k * 32;
    // A slice: 128 rows x 32 k (dim0 = K direction, contiguous)
    tdm_load_2d(A + (long)bm * 128 * K + kk, lds_addr_of(&a_sh[buf][0]),
                /*tile_x=*/32, /*tile_y=*/128, (unsigned long long)K,
                (unsigned long long)(K - kk),
                (unsigned long long)(Mr - bm * 128));
    // B slice: 32 k-rows x 64 n (dim0 = N direction, contiguous)
    tdm_load_2d(Bw + (long)kk * N + bn * 64, lds_addr_of(&b_sh[buf][0]),
                /*tile_x=*/64, /*tile_y=*/32, (unsigned long long)N,
                (unsigned long long)(N - bn * 64),
                (unsigned long long)(K - kk));
  };
  auto stage_fallback = [&](int k, int buf) {
    int kk = k * 32;
    for (int c = threadIdx.x; c < 512; c += 256) {  // A: 128x32 halfs
      int m = c >> 2, kq = (c & 3) * 8;
      *(uint4*)&a_sh[buf][m * 32 + kq] =
          *(const uint4*)&A[(long)(bm * 128 + m) * K + kk + kq];
    }
    {
      int c = threadIdx.x;  // B: 32x64 halfs = 256 chunks of 8
      int kr = c >> 3, nq = (c & 7) * 8;
      *(uint4*)&b_sh[buf][kr * 64 + nq] =
          *(const uint4*)&Bw[(long)(kk + kr) * N + bn * 64 + nq];
    }
  };

  v8f acc00 = v8f_zero(), acc01 = v8f_zero(), acc10 = v8f_zero(), acc11 = v8f_zero();

#if USE_TDM
  if (wave == 0) issue_tiles(0, 0);
#else
  stage_fallback(0, 0);
#endif

  for (int k = 0; k < nk; ++k) {
    int cur = k & 1;
#if USE_TDM
    if (wave == 0) {
      if (k + 1 < nk) {
        issue_tiles(k + 1, cur ^ 1);
        __builtin_amdgcn_s_wait_tensorcnt(2);  // step-k pair has landed
      } else {
        __builtin_amdgcn_s_wait_tensorcnt(0);
      }
    }
#else
    if (k + 1 < nk) stage_fallback(k + 1, cur ^ 1);
#endif
    __syncthreads();  // staging of buf[cur] visible to all waves
    const unsigned short* ac = &a_sh[cur][0];
    const unsigned short* bc = &b_sh[cur][0];
    v16bf a0 = load_a_frag(ac + (mw * 32) * 32, 32);
    v16bf a1 = load_a_frag(ac + (mw * 32 + 16) * 32, 32);
    v16bf b0 = load_b_frag(bc + nw * 32, 64);
    v16bf b1 = load_b_frag(bc + nw * 32 + 16, 64);
    acc00 = WMMA_BF16(a0, b0, acc00);
    acc01 = WMMA_BF16(a0, b1, acc01);
    acc10 = WMMA_BF16(a1, b0, acc10);
    acc11 = WMMA_BF16(a1, b1, acc11);
    __syncthreads();  // all waves done with buf[cur] before it is re-staged
  }

  int n = lane & 15, rb = (lane >> 4) << 3;
  v8f accs[2][2] = {{acc00, acc01}, {acc10, acc11}};
#pragma unroll
  for (int i = 0; i < 2; ++i) {
#pragma unroll
    for (int j = 0; j < 2; ++j) {
#pragma unroll
      for (int r = 0; r < 8; ++r) {
        long row = (long)bm * 128 + mw * 32 + i * 16 + rb + r;
        long col = (long)bn * 64 + nw * 32 + j * 16 + n;
        long idx = row * N + col;
        float v = accs[i][j][r];
        if (epi == 1) {
          C[idx] = v + extra[idx];
        } else if (epi == 2) {
          float g = extra[idx];
          float sg = g / (1.f + expf(-g));
          out_bf[idx] = f2bf(sg * v);
        } else {
          C[idx] = v;
        }
      }
    }
  }
}

// ---------------- RoPE + layout change to [B,H,S,HD] bf16 ----------------
__global__ __launch_bounds__(256) void rope_kernel(
    const float* __restrict__ qf, const float* __restrict__ kf,
    const float* __restrict__ vf, const int* __restrict__ pos_ids,
    unsigned short* __restrict__ qh, unsigned short* __restrict__ kh,
    unsigned short* __restrict__ vh) {
  int idx = blockIdx.x * blockDim.x + threadIdx.x;  // B*S*H*(HD/2) threads
  int j = idx & 63;
  int h = (idx >> 6) & (Hc - 1);
  int s = (idx >> 10) & (Sc - 1);
  int b = idx >> 20;
  long ibase = ((long)(b * Sc + s)) * Dc + (long)h * HDc;
  long obase = ((long)((b * Hc + h) * Sc + s)) * HDc;
  float p = (float)pos_ids[b * Sc + s];
  float inv = powf(ROPE_THETAc, -(float)(2 * j) / (float)HDc);
  float ang = p * inv;
  float c = cosf(ang), sn = sinf(ang);
  float q1 = qf[ibase + j], q2 = qf[ibase + j + 64];
  float k1 = kf[ibase + j], k2 = kf[ibase + j + 64];
  qh[obase + j]      = f2bf(q1 * c - q2 * sn);
  qh[obase + j + 64] = f2bf(q2 * c + q1 * sn);
  kh[obase + j]      = f2bf(k1 * c - k2 * sn);
  kh[obase + j + 64] = f2bf(k2 * c + k1 * sn);
  vh[obase + j]      = f2bf(vf[ibase + j]);
  vh[obase + j + 64] = f2bf(vf[ibase + j + 64]);
}

// ---------------- causal flash attention (one 16-row q tile per wave) -----
__global__ __launch_bounds__(256) void flash_attn_kernel(
    const unsigned short* __restrict__ qh, const unsigned short* __restrict__ kh,
    const unsigned short* __restrict__ vh, unsigned short* __restrict__ ctx_bf) {
  __shared__ unsigned short plds[8][16 * 32];
  int wave = threadIdx.x >> 5, lane = threadIdx.x & 31;
  int gid = (int)((blockIdx.x * blockDim.x + threadIdx.x) >> 5);
  int qt = gid & (Sc / 16 - 1);
  int h  = (gid >> 6) & (Hc - 1);
  int b  = gid >> 10;
  int q0 = qt * 16;
  int n = lane & 15, rb = (lane >> 4) << 3;
  long headbase = ((long)(b * Hc + h)) * Sc * HDc;

  v16bf aq[4];
#pragma unroll
  for (int kk = 0; kk < 4; ++kk)
    aq[kk] = load_a_frag(qh + headbase + (long)q0 * HDc + kk * 32, HDc);

  float m_run[8], l_run[8];
  v8f o[8];
#pragma unroll
  for (int r = 0; r < 8; ++r) { m_run[r] = -3e38f; l_run[r] = 0.f; }
#pragma unroll
  for (int c = 0; c < 8; ++c) o[c] = v8f_zero();

  const float scale = 0.08838834764831845f;  // 1/sqrt(128)
  int kend = q0 + 16;
  unsigned short* pw = &plds[wave][0];

  for (int kb = 0; kb < kend; kb += 32) {
    v8f sc0 = v8f_zero(), sc1 = v8f_zero();
    {
      const unsigned short* kbase = kh + headbase + (long)kb * HDc;
#pragma unroll
      for (int kk = 0; kk < 4; ++kk)
        sc0 = WMMA_BF16(aq[kk], load_bt_frag(kbase + kk * 32, HDc), sc0);
    }
    bool have1 = (kb + 16) < kend;  // wave-uniform
    if (have1) {
      const unsigned short* kbase = kh + headbase + (long)(kb + 16) * HDc;
#pragma unroll
      for (int kk = 0; kk < 4; ++kk)
        sc1 = WMMA_BF16(aq[kk], load_bt_frag(kbase + kk * 32, HDc), sc1);
    }
    float p0[8], p1[8];
#pragma unroll
    for (int r = 0; r < 8; ++r) {
      int qrow = q0 + rb + r;
      float v0 = (kb + n <= qrow) ? sc0[r] * scale : -3e38f;
      float v1 = (have1 && (kb + 16 + n <= qrow)) ? sc1[r] * scale : -3e38f;
      float mx = fmaxf(v0, v1);
      mx = fmaxf(mx, __shfl_xor(mx, 1, 32));
      mx = fmaxf(mx, __shfl_xor(mx, 2, 32));
      mx = fmaxf(mx, __shfl_xor(mx, 4, 32));
      mx = fmaxf(mx, __shfl_xor(mx, 8, 32));
      float m_new = fmaxf(m_run[r], mx);
      float corr = (m_run[r] > -1e30f) ? expf(m_run[r] - m_new) : 0.f;
      p0[r] = (v0 > -1e30f) ? expf(v0 - m_new) : 0.f;
      p1[r] = (v1 > -1e30f) ? expf(v1 - m_new) : 0.f;
      l_run[r] = l_run[r] * corr + p0[r] + p1[r];
      m_run[r] = m_new;
#pragma unroll
      for (int c = 0; c < 8; ++c) o[c][r] *= corr;
    }
#pragma unroll
    for (int r = 0; r < 8; ++r) {
      pw[(rb + r) * 32 + n]      = f2bf(p0[r]);
      pw[(rb + r) * 32 + 16 + n] = f2bf(p1[r]);
    }
    v16bf ap = load_a_frag(pw, 32);
#pragma unroll
    for (int c = 0; c < 8; ++c) {
      v16bf bv = load_b_frag(vh + headbase + (long)kb * HDc + c * 16, HDc);
      o[c] = WMMA_BF16(ap, bv, o[c]);
    }
  }
#pragma unroll
  for (int r = 0; r < 8; ++r) {
    float l = l_run[r];
    l += __shfl_xor(l, 1, 32);
    l += __shfl_xor(l, 2, 32);
    l += __shfl_xor(l, 4, 32);
    l += __shfl_xor(l, 8, 32);
    float rl = (l > 0.f) ? 1.f / l : 0.f;
    long row = (long)b * Sc + q0 + rb + r;
#pragma unroll
    for (int c = 0; c < 8; ++c)
      ctx_bf[row * Dc + (long)h * HDc + c * 16 + n] = f2bf(o[c][r] * rl);
  }
}

// ---------------- host-side orchestration ----------------
extern "C" void kernel_launch(void* const* d_in, const int* in_sizes, int n_in,
                              void* d_out, int out_size, void* d_ws, size_t ws_size,
                              hipStream_t stream) {
  (void)in_sizes; (void)n_in; (void)out_size; (void)ws_size;
  const float* hs     = (const float*)d_in[0];
  const int*   pos    = (const int*)d_in[1];
  const float* memb   = (const float*)d_in[2];
  const float* gatel  = (const float*)d_in[3];
  const float* wq     = (const float*)d_in[4];
  const float* wk     = (const float*)d_in[5];
  const float* wv     = (const float*)d_in[6];
  const float* wo     = (const float*)d_in[7];
  const float* wg     = (const float*)d_in[8];
  const float* wu     = (const float*)d_in[9];
  const float* wd     = (const float*)d_in[10];
  const float* rms1   = (const float*)d_in[11];
  const float* rms2   = (const float*)d_in[12];
  float* out = (float*)d_out;

  char* ws = (char*)d_ws;
  size_t off = 0;
  auto alloc = [&](size_t bytes) -> void* {
    void* p = (void*)(ws + off);
    off += (bytes + 255) & ~(size_t)255;
    return p;
  };
  const long BSD = BSc * Dc;
  const long DD  = (long)Dc * Dc;
  const long DF  = (long)Dc * Fc;
  const long PAD = 16 * HDc;

  unsigned short* hs_bf  = (unsigned short*)alloc(BSD * 2);
  unsigned short* mem_bf = (unsigned short*)alloc((long)Mc * Dc * 2);
  unsigned short* wq_bf  = (unsigned short*)alloc(DD * 2);
  unsigned short* wk_bf  = (unsigned short*)alloc(DD * 2);
  unsigned short* wv_bf  = (unsigned short*)alloc(DD * 2);
  unsigned short* wo_bf  = (unsigned short*)alloc(DD * 2);
  unsigned short* wg_bf  = (unsigned short*)alloc(DF * 2);
  unsigned short* wu_bf  = (unsigned short*)alloc(DF * 2);
  unsigned short* wd_bf  = (unsigned short*)alloc(DF * 2);
  int*            nni    = (int*)alloc(BSc * 4);
  float*          merged = (float*)alloc(BSD * 4);
  unsigned short* x_bf   = (unsigned short*)alloc(BSD * 2);
  float*          qf     = (float*)alloc(BSD * 4);
  float*          kf     = (float*)alloc(BSD * 4);
  float*          vf     = (float*)alloc(BSD * 4);
  unsigned short* qh     = (unsigned short*)alloc((BSD + PAD) * 2);
  unsigned short* kh     = (unsigned short*)alloc((BSD + PAD) * 2);
  unsigned short* vh     = (unsigned short*)alloc((BSD + PAD) * 2);
  unsigned short* ctx_bf = (unsigned short*)alloc(BSD * 2);
  float*          x2     = (float*)alloc(BSD * 4);
  unsigned short* y_bf   = (unsigned short*)alloc(BSD * 2);
  float*          gate_f = (float*)alloc(BSc * Fc * 4);
  unsigned short* act_bf = (unsigned short*)alloc(BSc * Fc * 2);

  auto cvt = [&](const float* s, unsigned short* d, long nelem) {
    int blocks = (int)((nelem + 2047) / 2048);
    cvt_bf16_kernel<<<blocks, 256, 0, stream>>>(s, d, nelem);
  };
  auto gemm = [&](const unsigned short* A, const unsigned short* Bw, float* C,
                  const float* extra, unsigned short* obf, int Mr, int K, int N,
                  int epi) {
    int blocks = (Mr / 128) * (N / 64);
    gemm_bf16_wmma_kernel<<<blocks, 256, 0, stream>>>(A, Bw, C, extra, obf, Mr, K, N, epi);
  };

  // 1) precision conversion
  cvt(hs, hs_bf, BSD);
  cvt(memb, mem_bf, (long)Mc * Dc);
  cvt(wq, wq_bf, DD); cvt(wk, wk_bf, DD); cvt(wv, wv_bf, DD); cvt(wo, wo_bf, DD);
  cvt(wg, wg_bf, DF); cvt(wu, wu_bf, DF); cvt(wd, wd_bf, DF);

  // 2) kNN retrieval (WMMA scores + argmax)
  knn_argmax_kernel<<<(int)(BSc / 16), 256, 0, stream>>>(hs_bf, mem_bf, nni);

  // 3) normalize + gated mix + RMSNorm-1
  mix_rms_kernel<<<(int)BSc, 256, 0, stream>>>(hs, memb, nni, gatel, rms1, merged, x_bf);

  // 4) QKV projections
  gemm(x_bf, wq_bf, qf, nullptr, nullptr, (int)BSc, Dc, Dc, 0);
  gemm(x_bf, wk_bf, kf, nullptr, nullptr, (int)BSc, Dc, Dc, 0);
  gemm(x_bf, wv_bf, vf, nullptr, nullptr, (int)BSc, Dc, Dc, 0);

  // 5) RoPE + [B,H,S,HD] bf16 layout
  rope_kernel<<<(Bc * Sc * Hc * (HDc / 2)) / 256, 256, 0, stream>>>(qf, kf, vf, pos, qh, kh, vh);

  // 6) causal flash attention
  flash_attn_kernel<<<(Bc * Hc * (Sc / 16)) / 8, 256, 0, stream>>>(qh, kh, vh, ctx_bf);

  // 7) output projection + residual -> x2
  gemm(ctx_bf, wo_bf, x2, merged, nullptr, (int)BSc, Dc, Dc, 1);

  // 8) RMSNorm-2
  rms2_kernel<<<(int)BSc, 256, 0, stream>>>(x2, rms2, y_bf);

  // 9) MLP: gate, up(+fused SiLU*mul), down(+fused residual into d_out)
  gemm(y_bf, wg_bf, gate_f, nullptr, nullptr, (int)BSc, Dc, Fc, 0);
  gemm(y_bf, wu_bf, gate_f, gate_f, act_bf, (int)BSc, Dc, Fc, 2);
  gemm(act_bf, wd_bf, out, x2, nullptr, (int)BSc, Fc, Dc, 1);
}
